// Net_18416819765870
// MI455X (gfx1250) — matrix operand order repaired and checked by reading, not compiled, not run on hardware
//
#include <hip/hip_runtime.h>
#include <hip/hip_bf16.h>

typedef __attribute__((ext_vector_type(16))) __bf16 v16bf;
typedef __attribute__((ext_vector_type(8)))  float  v8f;
typedef __attribute__((ext_vector_type(4)))  float  v4f;

#define T_STEPS 784
#define H       110
#define HP      128      // padded K: 110 real + K=110 bias row + K=111 x row + zeros
#define GP      512      // padded gate rows: 4 gates x 128
#define NB      16       // batch rows per workgroup
#define GST2    516      // gates [batch][row] stride in floats (2064B = 16 mod 256 -> conflict-free)

__device__ __forceinline__ float sigmoidf_(float v) {
    // v_exp_f32 + v_add + v_rcp_f32 (avoid IEEE div_scale/div_fmas chains)
    return __builtin_amdgcn_rcpf(1.0f + __expf(-v));
}
__device__ __forceinline__ float tanhf_(float v) {
#if __has_builtin(__builtin_amdgcn_tanhf)
    return __builtin_amdgcn_tanhf(v);        // gfx1250 v_tanh_f32
#else
    float e = __expf(2.0f * v);
    return (e - 1.0f) * __builtin_amdgcn_rcpf(e + 1.0f);
#endif
}

__global__ void zero_tail_kernel(float* out) { out[2560] = 0.0f; }

__global__ __launch_bounds__(256, 1)
void skiplstm_kernel(const float* __restrict__ x,
                     const float* __restrict__ W_ih,
                     const float* __restrict__ W_hh,
                     const float* __restrict__ bvec,
                     const float* __restrict__ W_p,
                     const float* __restrict__ b_p,
                     const float* __restrict__ h0,
                     const float* __restrict__ c0,
                     const float* __restrict__ W_fc,
                     const float* __restrict__ b_fc,
                     float* __restrict__ out)
{
    __shared__ __align__(16) float  gT[NB * GST2];      // gates [16 batch][516] f32
    __shared__ __align__(32) __bf16 hT[HP * NB];        // h^T bf16 [128][16]; row110=1, row111=x_t
    __shared__ float  wp[H];
    __shared__ float  u_s[NB], ub_s[NB], duacc[NB];
    __shared__ float  bp_s;

    const int tid  = threadIdx.x;
    const int wave = tid >> 5;
    const int lane = tid & 31;
    const int gb0  = blockIdx.x * NB;   // first global batch row of this WG

    // ---------------- init LDS state ----------------
    for (int idx = tid; idx < HP * NB; idx += 256) {
        int j = idx >> 4, b = idx & 15;
        float v;
        if      (j <  H)     v = h0[j];
        else if (j == H)     v = 1.0f;                       // bias row
        else if (j == H + 1) v = x[(gb0 + b) * T_STEPS + 0]; // x row (t=0)
        else                 v = 0.0f;
        hT[idx] = (__bf16)v;
    }
    for (int j = tid; j < H; j += 256) wp[j] = W_p[j];
    if (tid < NB) { u_s[tid] = 1.0f; ub_s[tid] = 1.0f; duacc[tid] = 0.0f; }
    if (tid == 0) bp_s = b_p[0];

    // ------- loop-invariant A fragments (bf16): W_hh | bias (K=110) | W_ih (K=111) -------
    // A 16x32 layout: lanes 0-15 -> M=lane, Kbase 0; lanes 16-31 -> M=lane-16, Kbase 8/24.
    v16bf Af[4][4];
    {
        const int kb0 = (lane < 16) ? 0 : 8;
        const int m   = lane & 15;
        #pragma unroll
        for (int mtl = 0; mtl < 4; ++mtl) {
            int R = (wave * 4 + mtl) * 16 + m;      // padded gate row
            int q = R >> 7, jr = R & 127;
            bool rowok = (jr < H);
            int orow = q * H + jr;
            #pragma unroll
            for (int kc = 0; kc < 4; ++kc) {
                v16bf a;
                #pragma unroll
                for (int e = 0; e < 16; ++e) {
                    int vg = e >> 1, p = e & 1;
                    int k  = (vg < 4) ? (kb0 + 2 * vg + p)
                                      : (16 + kb0 + 2 * (vg - 4) + p);
                    int kg = kc * 32 + k;
                    float w = 0.0f;
                    if (rowok) {
                        if      (kg <  H)     w = W_hh[orow * H + kg];
                        else if (kg == H)     w = bvec[orow];
                        else if (kg == H + 1) w = W_ih[orow];
                    }
                    a[e] = (__bf16)w;
                }
                Af[mtl][kc] = a;
            }
        }
    }

    // ------- per-thread recurrent state (fixed (j,b) ownership across all steps) -------
    const int pb = tid & 15;        // batch column owned in phase B
    const int jb = tid >> 4;
    float creg[7], hreg[7];
    #pragma unroll
    for (int it = 0; it < 7; ++it) {
        int j = it * 16 + jb;
        creg[it] = (j < H) ? c0[j] : 0.0f;
        hreg[it] = (j < H) ? h0[j] : 0.0f;
    }

    float tot = 0.0f;   // wave0 lanes<16: running sum of ub (exact small ints)

    __syncthreads();

    for (int t = 0; t < T_STEPS; ++t) {
        // ---- phase A: gates^T = [W_hh|b|W_ih] . [h;1;x]^T  (C starts at zero) ----
        const int bb   = lane & 15;
        const int rofs = (lane < 16) ? 0 : 8;

        v16bf bfr[4];
        #pragma unroll
        for (int kc = 0; kc < 4; ++kc) {
            #pragma unroll
            for (int s = 0; s < 16; ++s) {
                int k = (s < 8) ? (rofs + s) : (16 + rofs + (s - 8));
                bfr[kc][s] = hT[(kc * 32 + k) * NB + bb];
            }
        }

        v8f acc[4] = {};
        #pragma unroll
        for (int kc = 0; kc < 4; ++kc) {
            #pragma unroll
            for (int mtl = 0; mtl < 4; ++mtl) {
                acc[mtl] = __builtin_amdgcn_wmma_f32_16x16x32_bf16(
                    false, Af[mtl][kc], false, bfr[kc], (short)0, acc[mtl],
                    false, false);
            }
        }
        #pragma unroll
        for (int mtl = 0; mtl < 4; ++mtl) {
            int Rb = (wave * 4 + mtl) * 16 + rofs;   // 8 consecutive padded rows
            v4f lo = __builtin_shufflevector(acc[mtl], acc[mtl], 0, 1, 2, 3);
            v4f hi = __builtin_shufflevector(acc[mtl], acc[mtl], 4, 5, 6, 7);
            *(v4f*)&gT[bb * GST2 + Rb]     = lo;
            *(v4f*)&gT[bb * GST2 + Rb + 4] = hi;
        }
        __syncthreads();

        // ---- phase B: elementwise LSTM + skip blend; also stage next x into hT ----
        {
            float ubc  = ub_s[pb];
            float omu  = 1.0f - ubc;
            float dpart = 0.0f;
            const float* g0 = &gT[pb * GST2];
            #pragma unroll
            for (int it = 0; it < 7; ++it) {
                int j = it * 16 + jb;
                if (j < H) {
                    float gi = sigmoidf_(g0[j]);
                    float gf = sigmoidf_(g0[128 + j]);
                    float gg = tanhf_   (g0[256 + j]);
                    float go = sigmoidf_(g0[384 + j]);
                    float cnew = gf * creg[it] + gi * gg;
                    float hnew = go * tanhf_(cnew);
                    float cn = ubc * cnew + omu * creg[it];
                    float hn = ubc * hnew + omu * hreg[it];
                    creg[it] = cn;
                    hreg[it] = hn;
                    hT[j * NB + pb] = (__bf16)hn;
                    dpart += cn * wp[j];
                }
            }
            atomicAdd(&duacc[pb], dpart);   // ds_add_f32
            // x row for next step (owned by wave0 lanes 0-15; row 111 untouched above)
            if (tid < NB && t + 1 < T_STEPS) {
                hT[(H + 1) * NB + pb] = (__bf16)x[(gb0 + pb) * T_STEPS + t + 1];
                if (t + 2 < T_STEPS)
                    __builtin_prefetch(&x[(gb0 + pb) * T_STEPS + t + 2], 0, 0);
            }
        }
        __syncthreads();

        // ---- phase C (wave 0 only, no trailing barrier): skip-gate state update ----
        // Writes (u_s, ub_s, duacc) are first consumed by next phase B, which is
        // already ordered after this by the A->B barrier; wave 0 itself runs C
        // before its own phase A of t+1.
        if (wave == 0 && lane < NB) {
            int b = lane;
            float du  = sigmoidf_(duacc[b] + bp_s);
            float uc  = u_s[b];
            float ubc = ub_s[b];
            tot += ubc;
            float un = (ubc > 0.5f) ? du : (uc + fminf(du, 1.0f - uc));
            u_s[b]  = un;
            ub_s[b] = (un >= 0.5f) ? 1.0f : 0.0f;
            duacc[b] = 0.0f;
        }
    }

    // ---- final FC on last h (16 rows x 10 classes per WG) ----
    if (tid < NB * 10) {
        int b = tid / 10, k = tid % 10;
        float s = b_fc[k];
        for (int j = 0; j < H; ++j)
            s += (float)hT[j * NB + b] * W_fc[k * H + j];
        out[(gb0 + b) * 10 + k] = s;
    }

    // ---- total_u: exact integer sums -> deterministic float atomics ----
    if (wave == 0 && lane < NB) duacc[lane] = tot;
    __syncthreads();
    if (tid == 0) {
        float s = 0.0f;
        for (int i = 0; i < NB; ++i) s += duacc[i];
        atomicAdd(&out[2560], s);
    }
}

extern "C" void kernel_launch(void* const* d_in, const int* in_sizes, int n_in,
                              void* d_out, int out_size, void* d_ws, size_t ws_size,
                              hipStream_t stream) {
    const float* x    = (const float*)d_in[0];
    const float* W_ih = (const float*)d_in[1];
    const float* W_hh = (const float*)d_in[2];
    const float* b    = (const float*)d_in[3];
    const float* W_p  = (const float*)d_in[4];
    const float* b_p  = (const float*)d_in[5];
    const float* h0   = (const float*)d_in[6];
    const float* c0   = (const float*)d_in[7];
    const float* W_fc = (const float*)d_in[8];
    const float* b_fc = (const float*)d_in[9];
    float* out = (float*)d_out;

    zero_tail_kernel<<<1, 1, 0, stream>>>(out);
    skiplstm_kernel<<<16, 256, 0, stream>>>(x, W_ih, W_hh, b, W_p, b_p,
                                            h0, c0, W_fc, b_fc, out);
}
